// EncoderLayer_75024488726611
// MI455X (gfx1250) — matrix operand order repaired and checked by reading, not compiled, not run on hardware
//
#include <hip/hip_runtime.h>
#include <stdint.h>

// ---------------------------------------------------------------------------
// CDNA5 (gfx1250) types and helpers
// ---------------------------------------------------------------------------
typedef __attribute__((ext_vector_type(16))) __bf16      v16bf;
typedef __attribute__((ext_vector_type(8)))  float       v8f;
typedef __attribute__((ext_vector_type(4)))  unsigned int v4u;

union Frag { v4u u[2]; v16bf v; };

// pack two f32 -> packed bf16x2 (low = a, high = b).
// Prefer v_cvt_pk_bf16_f32 if the toolchain has it; else round-half-up +
// one v_perm_b32 (3 VALU per pair).
__device__ __forceinline__ unsigned int pack_bf16(float a, float b) {
#if defined(__has_builtin) && __has_builtin(__builtin_amdgcn_cvt_pk_bf16_f32)
  auto p = __builtin_amdgcn_cvt_pk_bf16_f32(a, b);
  union U { decltype(p) v; unsigned int u; } c;
  c.u = 0u; c.v = p;
  return c.u;
#else
  unsigned int ua = __float_as_uint(a) + 0x8000u;
  unsigned int ub = __float_as_uint(b) + 0x8000u;
  return __builtin_amdgcn_perm(ub, ua, 0x07060302u);  // {ub[31:16], ua[31:16]}
#endif
}

__device__ __forceinline__ unsigned short f2bf(float a) {
  return (unsigned short)((__float_as_uint(a) + 0x8000u) >> 16);
}

__device__ __forceinline__ v8f wmma_bf16(v16bf a, v16bf b, v8f c) {
  return __builtin_amdgcn_wmma_f32_16x16x32_bf16(
      false, a, false, b, (short)0, c, false, false);
}

// ---------------------------------------------------------------------------
// Tiled GEMM: out = act( scale*(X[M,K] * W[K,N] + bias[N]) (+ residual[M,N]) )
// Output either f32 (Yf) or packed bf16 (Y16, when non-null).
// 128x128 block tile, BK=32, 256 threads = 8 waves, 2x4 WMMA subtiles/wave.
// Double-buffered LDS: one barrier per K-chunk; next tile's global loads
// issue before the WMMA block and the pack/ds_store lands after it.
// ---------------------------------------------------------------------------
__global__ __launch_bounds__(256, 2) void gemm_bf16_wmma(
    const float* __restrict__ X, const float* __restrict__ W,
    const float* __restrict__ bias, const float* __restrict__ residual,
    float* __restrict__ Yf, unsigned short* __restrict__ Y16,
    int M, int N, int K, int relu, float scale)
{
  __shared__ __align__(16) unsigned short As[2][128][32];  // [m][k] bf16
  __shared__ __align__(16) unsigned short Bs[2][128][40];  // [n][k] bf16 (pad)

  const int t    = threadIdx.x;
  const int wave = t >> 5, lane = t & 31;
  const int half = lane >> 4, lm = lane & 15;
  const int waveM = (wave >> 1) * 32;
  const int waveN = (wave & 1)  * 64;
  const int m0 = blockIdx.y * 128;
  const int n0 = blockIdx.x * 128;

  v8f acc[2][4];
#pragma unroll
  for (int mi = 0; mi < 2; ++mi)
#pragma unroll
    for (int ni = 0; ni < 4; ++ni) acc[mi][ni] = (v8f){0,0,0,0,0,0,0,0};

  // loader mappings
  const int arow = t >> 1,        acol = (t & 1) * 16;   // A: 16 k-contig floats
  const int bk2  = (t >> 4) * 2,  bnc  = (t & 15) * 8;   // B: 2 k-rows x 8 n-cols

  float4 a4[4], b4[4];

  auto loadg = [&](int k0) {
    const float* asrc = X + (size_t)(m0 + arow) * K + (k0 + acol);
    a4[0] = *(const float4*)(asrc + 0);
    a4[1] = *(const float4*)(asrc + 4);
    a4[2] = *(const float4*)(asrc + 8);
    a4[3] = *(const float4*)(asrc + 12);
    const float* bsrc = W + (size_t)(k0 + bk2) * N + (n0 + bnc);
    b4[0] = *(const float4*)(bsrc);
    b4[1] = *(const float4*)(bsrc + 4);
    b4[2] = *(const float4*)(bsrc + N);
    b4[3] = *(const float4*)(bsrc + N + 4);
  };

  auto stage = [&](int pb) {
    v4u pa0 = (v4u){ pack_bf16(a4[0].x, a4[0].y), pack_bf16(a4[0].z, a4[0].w),
                     pack_bf16(a4[1].x, a4[1].y), pack_bf16(a4[1].z, a4[1].w) };
    v4u pa1 = (v4u){ pack_bf16(a4[2].x, a4[2].y), pack_bf16(a4[2].z, a4[2].w),
                     pack_bf16(a4[3].x, a4[3].y), pack_bf16(a4[3].z, a4[3].w) };
    *(v4u*)&As[pb][arow][acol]     = pa0;
    *(v4u*)&As[pb][arow][acol + 8] = pa1;
    float bl[8] = { b4[0].x, b4[0].y, b4[0].z, b4[0].w,
                    b4[1].x, b4[1].y, b4[1].z, b4[1].w };
    float bh[8] = { b4[2].x, b4[2].y, b4[2].z, b4[2].w,
                    b4[3].x, b4[3].y, b4[3].z, b4[3].w };
#pragma unroll
    for (int j = 0; j < 8; ++j)
      *(unsigned int*)&Bs[pb][bnc + j][bk2] = pack_bf16(bl[j], bh[j]);
  };

  loadg(0);
  stage(0);
  int pb = 0;
  for (int k0 = 0; k0 < K; k0 += 32, pb ^= 1) {
    __syncthreads();                    // buffer pb now visible to all waves
    const bool more = (k0 + 32 < K);
    if (more) loadg(k0 + 32);           // overlap global loads with WMMA

    Frag a[2];
#pragma unroll
    for (int mi = 0; mi < 2; ++mi) {
      int m = waveM + mi * 16 + lm;
      a[mi].u[0] = *(const v4u*)&As[pb][m][half * 8];
      a[mi].u[1] = *(const v4u*)&As[pb][m][16 + half * 8];
    }
#pragma unroll
    for (int ni = 0; ni < 4; ++ni) {
      Frag bfr;
      int n = waveN + ni * 16 + lm;
      bfr.u[0] = *(const v4u*)&Bs[pb][n][half * 8];
      bfr.u[1] = *(const v4u*)&Bs[pb][n][16 + half * 8];
#pragma unroll
      for (int mi = 0; mi < 2; ++mi)
        acc[mi][ni] = wmma_bf16(a[mi].v, bfr.v, acc[mi][ni]);
    }

    if (more) stage(pb ^ 1);            // pack+store into the other buffer
  }

  // epilogue: C layout row = i + 8*half, col = lane 0..15
#pragma unroll
  for (int ni = 0; ni < 4; ++ni) {
    int col = n0 + waveN + ni * 16 + lm;
    float bv = bias[col];
#pragma unroll
    for (int mi = 0; mi < 2; ++mi) {
#pragma unroll
      for (int i = 0; i < 8; ++i) {
        int row = m0 + waveM + mi * 16 + half * 8 + i;
        float val = (acc[mi][ni][i] + bv) * scale;
        if (residual) val += residual[(size_t)row * N + col];
        if (relu)     val = fmaxf(val, 0.0f);
        if (Y16) Y16[(size_t)row * N + col] = f2bf(val);
        else     Yf [(size_t)row * N + col] = val;
      }
    }
  }
}

// ---------------------------------------------------------------------------
// Flash attention over bf16 Q/K/V (Q pre-scaled by 1/sqrt(64) in projection).
// Grid: B*H*(S/128) blocks of 256 threads (8 waves); wave owns 16 q rows.
// Double-buffered K/V staging -> one __syncthreads per 32-key chunk; the
// P relayout tile is wave-private (same-wave LDS ordering suffices).
// ---------------------------------------------------------------------------
__global__ __launch_bounds__(256, 1) void attn_flash(
    const unsigned short* __restrict__ Qp, const unsigned short* __restrict__ Kp,
    const unsigned short* __restrict__ Vp, const int* __restrict__ mask,
    float* __restrict__ ctx)
{
  const int S = 4096, D = 768, H = 12;
  const int qb = blockIdx.x & 31;     // S/128 query blocks
  const int bh = blockIdx.x >> 5;
  const int b  = bh / H, h = bh % H;

  __shared__ __align__(16) unsigned short Qs[128][72];      // [qrow][dk]
  __shared__ __align__(16) unsigned short Ks[2][32][72];    // [key][dk]
  __shared__ __align__(16) unsigned short Vs[2][64][40];    // [dk][key]
  __shared__ __align__(16) unsigned short Ps[8][16][32];    // per-wave P tile

  const int t = threadIdx.x, wave = t >> 5, lane = t & 31;
  const int half = lane >> 4, lm = lane & 15;
  const size_t headoff = (size_t)b * S * D + (size_t)h * 64;

  // ---- load Q block (pure bf16 copy, 64B per thread) ----
  {
    int row = t >> 1, ch = (t & 1) * 32;
    const v4u* src = (const v4u*)(Qp + headoff + (size_t)(qb * 128 + row) * D + ch);
    *(v4u*)&Qs[row][ch]      = src[0];
    *(v4u*)&Qs[row][ch + 8]  = src[1];
    *(v4u*)&Qs[row][ch + 16] = src[2];
    *(v4u*)&Qs[row][ch + 24] = src[3];
  }
  __syncthreads();

  // ---- Q as two A-fragments (dk 0..31, 32..63), kept in registers ----
  Frag qf[2];
  {
    int qr = wave * 16 + lm;
#pragma unroll
    for (int c = 0; c < 2; ++c) {
      qf[c].u[0] = *(const v4u*)&Qs[qr][c * 32 + half * 8];
      qf[c].u[1] = *(const v4u*)&Qs[qr][c * 32 + 16 + half * 8];
    }
  }

  float mr[8], lr[8];
#pragma unroll
  for (int i = 0; i < 8; ++i) { mr[i] = -3.0e38f; lr[i] = 0.0f; }
  v8f o[4];
#pragma unroll
  for (int ni = 0; ni < 4; ++ni) o[ni] = (v8f){0,0,0,0,0,0,0,0};

  const int kvr  = t >> 3,        kvc8 = (t & 7) * 8;   // K loader: 16B copy
  const int key2 = (t & 15) * 2,  dkc  = (t >> 4) * 4;  // V loader: perm repack

  v4u  kreg;
  uint2 va, vc;
  auto loadkv = [&](int kc) {
    kreg = *(const v4u*)(Kp + headoff + (size_t)(kc + kvr) * D + kvc8);
    const unsigned short* r0 = Vp + headoff + (size_t)(kc + key2) * D + dkc;
    va = *(const uint2*)r0;
    vc = *(const uint2*)(r0 + D);
  };
  auto stagekv = [&](int pbuf) {
    *(v4u*)&Ks[pbuf][kvr][kvc8] = kreg;
    *(unsigned int*)&Vs[pbuf][dkc + 0][key2] = __builtin_amdgcn_perm(vc.x, va.x, 0x05040100u);
    *(unsigned int*)&Vs[pbuf][dkc + 1][key2] = __builtin_amdgcn_perm(vc.x, va.x, 0x07060302u);
    *(unsigned int*)&Vs[pbuf][dkc + 2][key2] = __builtin_amdgcn_perm(vc.y, va.y, 0x05040100u);
    *(unsigned int*)&Vs[pbuf][dkc + 3][key2] = __builtin_amdgcn_perm(vc.y, va.y, 0x07060302u);
  };

  loadkv(0);
  stagekv(0);
  int pb = 0;
  for (int kc = 0; kc < S; kc += 32, pb ^= 1) {
    __syncthreads();                    // K/V buffer pb visible
    const bool more = (kc + 32 < S);
    if (more) loadkv(kc + 32);          // overlap next chunk's global loads

    // ---- scores for 32 keys: two 16-key C tiles, accumulate over dk ----
    v8f s0 = (v8f){0,0,0,0,0,0,0,0}, s1 = (v8f){0,0,0,0,0,0,0,0};
#pragma unroll
    for (int c = 0; c < 2; ++c) {
      Frag kf0, kf1;
      kf0.u[0] = *(const v4u*)&Ks[pb][lm][c * 32 + half * 8];
      kf0.u[1] = *(const v4u*)&Ks[pb][lm][c * 32 + 16 + half * 8];
      kf1.u[0] = *(const v4u*)&Ks[pb][16 + lm][c * 32 + half * 8];
      kf1.u[1] = *(const v4u*)&Ks[pb][16 + lm][c * 32 + 16 + half * 8];
      s0 = wmma_bf16(qf[c].v, kf0.v, s0);
      s1 = wmma_bf16(qf[c].v, kf1.v, s1);
    }

    // ---- mask ----
    int mk0 = mask[b * S + kc + lm];
    int mk1 = mask[b * S + kc + 16 + lm];
#pragma unroll
    for (int i = 0; i < 8; ++i) {
      if (mk0 == 0) s0[i] = -3.0e38f;
      if (mk1 == 0) s1[i] = -3.0e38f;
    }

    // ---- online softmax; rows in elements, keys across 16-lane halves ----
    float tm[8];
#pragma unroll
    for (int i = 0; i < 8; ++i) tm[i] = fmaxf(s0[i], s1[i]);
#pragma unroll
    for (int off = 1; off < 16; off <<= 1)
#pragma unroll
      for (int i = 0; i < 8; ++i)
        tm[i] = fmaxf(tm[i], __shfl_xor(tm[i], off, 32));

    float alpha[8], p0[8], p1[8], rs[8];
#pragma unroll
    for (int i = 0; i < 8; ++i) {
      float mn = fmaxf(mr[i], tm[i]);
      alpha[i] = __expf(mr[i] - mn);
      mr[i] = mn;
      p0[i] = __expf(s0[i] - mn);
      p1[i] = __expf(s1[i] - mn);
      rs[i] = p0[i] + p1[i];
    }
#pragma unroll
    for (int off = 1; off < 16; off <<= 1)
#pragma unroll
      for (int i = 0; i < 8; ++i) rs[i] += __shfl_xor(rs[i], off, 32);
#pragma unroll
    for (int i = 0; i < 8; ++i) lr[i] = lr[i] * alpha[i] + rs[i];
#pragma unroll
    for (int ni = 0; ni < 4; ++ni)
#pragma unroll
      for (int i = 0; i < 8; ++i) o[ni][i] *= alpha[i];

    // ---- C-layout -> A-layout relayout of P (wave-private LDS tile; LDS
    //      ops from one wave are in-order, so no workgroup barrier) ----
#pragma unroll
    for (int i = 0; i < 8; ++i) {
      Ps[wave][half * 8 + i][lm]      = f2bf(p0[i]);
      Ps[wave][half * 8 + i][16 + lm] = f2bf(p1[i]);
    }
    __builtin_amdgcn_wave_barrier();

    Frag pf;
    pf.u[0] = *(const v4u*)&Ps[wave][lm][half * 8];
    pf.u[1] = *(const v4u*)&Ps[wave][lm][16 + half * 8];
#pragma unroll
    for (int ni = 0; ni < 4; ++ni) {
      Frag vf;
      int dk = ni * 16 + lm;
      vf.u[0] = *(const v4u*)&Vs[pb][dk][half * 8];
      vf.u[1] = *(const v4u*)&Vs[pb][dk][16 + half * 8];
      o[ni] = wmma_bf16(pf.v, vf.v, o[ni]);
    }

    if (more) stagekv(pb ^ 1);          // stage next chunk into other buffer
  }

  // ---- normalize and store ctx (f32) in [B,S,D] layout ----
#pragma unroll
  for (int ni = 0; ni < 4; ++ni)
#pragma unroll
    for (int i = 0; i < 8; ++i) {
      int row = qb * 128 + wave * 16 + half * 8 + i;
      int col = h * 64 + ni * 16 + lm;
      ctx[(size_t)b * S * D + (size_t)row * D + col] = o[ni][i] / lr[i];
    }
}

// ---------------------------------------------------------------------------
// Row LayerNorm over D=768. One block (256 threads) per row, 3 elems/thread.
// ---------------------------------------------------------------------------
__global__ __launch_bounds__(256) void ln_kernel(
    const float* __restrict__ x, const float* __restrict__ g,
    const float* __restrict__ be, float* __restrict__ y)
{
  const int Dn = 768;
  const int row = blockIdx.x, t = threadIdx.x;
  const float* xr = x + (size_t)row * Dn;
  float v0 = xr[t], v1 = xr[t + 256], v2 = xr[t + 512];

  __shared__ float red[8];
  float s = v0 + v1 + v2;
#pragma unroll
  for (int off = 16; off; off >>= 1) s += __shfl_xor(s, off, 32);
  if ((t & 31) == 0) red[t >> 5] = s;
  __syncthreads();
  float tot = 0.0f;
#pragma unroll
  for (int i = 0; i < 8; ++i) tot += red[i];
  float mean = tot * (1.0f / 768.0f);
  __syncthreads();

  float d0 = v0 - mean, d1 = v1 - mean, d2 = v2 - mean;
  float q = d0 * d0 + d1 * d1 + d2 * d2;
#pragma unroll
  for (int off = 16; off; off >>= 1) q += __shfl_xor(q, off, 32);
  if ((t & 31) == 0) red[t >> 5] = q;
  __syncthreads();
  float tq = 0.0f;
#pragma unroll
  for (int i = 0; i < 8; ++i) tq += red[i];
  float rstd = rsqrtf(tq * (1.0f / 768.0f) + 1e-6f);

  float* yr = y + (size_t)row * Dn;
  yr[t]       = d0 * rstd * g[t]       + be[t];
  yr[t + 256] = d1 * rstd * g[t + 256] + be[t + 256];
  yr[t + 512] = d2 * rstd * g[t + 512] + be[t + 512];
}

// ---------------------------------------------------------------------------
// Orchestration
// ---------------------------------------------------------------------------
extern "C" void kernel_launch(void* const* d_in, const int* in_sizes, int n_in,
                              void* d_out, int out_size, void* d_ws, size_t ws_size,
                              hipStream_t stream)
{
  (void)in_sizes; (void)n_in; (void)out_size; (void)ws_size;

  const float* inq  = (const float*)d_in[0];
  const float* ink  = (const float*)d_in[1];
  const float* inv  = (const float*)d_in[2];
  const int*   mask = (const int*)d_in[3];
  const float* wq = (const float*)d_in[4];  const float* bq = (const float*)d_in[5];
  const float* wk = (const float*)d_in[6];  const float* bk = (const float*)d_in[7];
  const float* wv = (const float*)d_in[8];  const float* bv = (const float*)d_in[9];
  const float* wo = (const float*)d_in[10]; const float* bo = (const float*)d_in[11];
  const float* w1 = (const float*)d_in[12]; const float* b1 = (const float*)d_in[13];
  const float* w2 = (const float*)d_in[14]; const float* b2 = (const float*)d_in[15];
  const float* g1 = (const float*)d_in[16]; const float* be1 = (const float*)d_in[17];
  const float* g2 = (const float*)d_in[18]; const float* be2 = (const float*)d_in[19];

  const int M = 2 * 4096, D = 768, F = 3072;
  const size_t SZ = (size_t)M * D;           // elements per [M,D] buffer

  char* w = (char*)d_ws;
  unsigned short* Qb = (unsigned short*)w;             // bf16 [M,D]
  unsigned short* Kb = Qb + SZ;
  unsigned short* Vb = Kb + SZ;
  float* f0 = (float*)(w + 3 * SZ * sizeof(unsigned short)); // ctx, later preLN2
  float* f1 = f0 + SZ;                                 // preLN1
  float* f2 = f1 + SZ;                                 // x1
  float* f3 = f2 + SZ;                                 // FFN hidden [M,F]

  dim3 blk(256);
  dim3 gDD(D / 128, M / 128);                // (6, 64)
  dim3 gDF(F / 128, M / 128);                // (24, 64)

  // QKV projections -> bf16 (Q pre-scaled by 1/sqrt(DK))
  gemm_bf16_wmma<<<gDD, blk, 0, stream>>>(inq, wq, bq, nullptr, nullptr, Qb,
                                          M, D, D, 0, 0.125f);
  gemm_bf16_wmma<<<gDD, blk, 0, stream>>>(ink, wk, bk, nullptr, nullptr, Kb,
                                          M, D, D, 0, 1.0f);
  gemm_bf16_wmma<<<gDD, blk, 0, stream>>>(inv, wv, bv, nullptr, nullptr, Vb,
                                          M, D, D, 0, 1.0f);

  // flash attention -> ctx (f32)
  attn_flash<<<dim3(2 * 12 * 32), blk, 0, stream>>>(Qb, Kb, Vb, mask, f0);

  // output projection + residual(inputs_q), then LN1 -> x1
  gemm_bf16_wmma<<<gDD, blk, 0, stream>>>(f0, wo, bo, inq, f1, nullptr,
                                          M, D, D, 0, 1.0f);
  ln_kernel<<<dim3(M), blk, 0, stream>>>(f1, g1, be1, f2);

  // FFN: relu(x1@w1+b1) @ w2 + b2 + residual(x1), then LN2 -> out
  gemm_bf16_wmma<<<gDF, blk, 0, stream>>>(f2, w1, b1, nullptr, f3, nullptr,
                                          M, F, D, 1, 1.0f);
  gemm_bf16_wmma<<<gDD, blk, 0, stream>>>(f3, w2, b2, f2, f0, nullptr,
                                          M, D, F, 0, 1.0f);
  ln_kernel<<<dim3(M), blk, 0, stream>>>(f0, g2, be2, (float*)d_out);
}
